// Decoder_7550552507001
// MI455X (gfx1250) — matrix operand order repaired and checked by reading, not compiled, not run on hardware
//
#include <hip/hip_runtime.h>
#include <hip/hip_bf16.h>
#include <math.h>

// ---------------------------------------------------------------- constants
#define BATCH 2
#define SEQ   2048
#define DMODEL 1024
#define NHEAD 16
#define DK    64
#define DFF   4096
#define MROWS (BATCH * SEQ)          // 4096
#define NEGV  (-10000.0f)

typedef __bf16 bf16;
typedef __attribute__((ext_vector_type(16))) __bf16 v16bf;
typedef __attribute__((ext_vector_type(8)))  float  v8f;
typedef __attribute__((ext_vector_type(4)))  float  f32x4;

union FragBF {           // one WMMA 16x16x32 bf16 operand (16 bf16 per lane)
    v16bf v;
    f32x4 f[2];          // two contiguous 8-element (16B) chunks
    bf16  e[16];
};

__device__ inline float gelu_exact(float x) {
    return 0.5f * x * (1.0f + erff(x * 0.70710678118654752440f));
}

// ---- CDNA5 async global->LDS copy (ASYNCcnt-tracked, no VGPR round trip)
__device__ inline unsigned int lds_byte_off(const void* p) {
    // generic pointer to LDS: addr[63:32] = shared aperture, addr[31:0] = LDS offset
    return (unsigned int)(unsigned long long)p;
}

__device__ inline void async_copy_b128(unsigned int lds_off, const void* gptr) {
    // GLOBAL_LOAD_ASYNC_TO_LDS_B128 (VGLOBAL op 98): VDST = LDS byte address,
    // VADDR = 64-bit global address ("GV" mode). Tracked by ASYNCcnt.
    asm volatile("global_load_async_to_lds_b128 %0, %1, off"
                 :: "v"(lds_off), "v"(gptr) : "memory");
}

__device__ inline void wait_async0() {
    asm volatile("s_wait_asynccnt 0" ::: "memory");
}

// ---------------------------------------------------------------- f32 -> bf16
__global__ __launch_bounds__(256) void cvt_f32_bf16(const float* __restrict__ x,
                                                    bf16* __restrict__ y, long n) {
    long i = (long)blockIdx.x * 256 + threadIdx.x;
    long stride = (long)gridDim.x * 256;
    for (; i < n; i += stride) y[i] = (bf16)x[i];
}

// ---------------------------------------------------------------- GEMM
// C[M,N] = act( (A[M,K] @ W[N,K]^T + bias[N]) * scale )   (bf16 in, f32 accum)
// Block: 256 threads (8 waves). Block tile 128(M) x 128(N), BK = 32,
// double-buffered LDS filled with GLOBAL_LOAD_ASYNC_TO_LDS_B128.
// Waves: 4(M) x 2(N); each wave 32x64 = 2x4 grid of v_wmma_f32_16x16x32_bf16.
template <bool GELU, bool OUTBF>
__global__ __launch_bounds__(256)
void gemm_bf16_nt(const bf16* __restrict__ A, const bf16* __restrict__ W,
                  const float* __restrict__ bias, const float* __restrict__ scale_ptr,
                  float scale_mul, float* __restrict__ Cf, bf16* __restrict__ Cb,
                  int M, int N, int K) {
    __shared__ bf16 As[2][128][40];   // +8 pad keeps 16B row alignment (80B stride)
    __shared__ bf16 Bs[2][128][40];

    const int tid  = threadIdx.x;
    const int wave = tid >> 5;
    const int lane = tid & 31;
    const int ln   = lane & 15;
    const int lh   = lane >> 4;
    const int m0   = blockIdx.y * 128;
    const int n0   = blockIdx.x * 128;
    const int wm   = (wave & 3) * 32;     // 4 waves along M
    const int wn   = (wave >> 2) * 64;    // 2 waves along N

    // staging coords: each thread moves 2x16B for A and 2x16B for B per stage
    const int lr  = tid >> 1;             // 0..127 (tile row)
    const int lsg = (tid & 1) * 16;       // 0 or 16 (k segment)
    const bf16* gA = A + (size_t)(m0 + lr) * K + lsg;
    const bf16* gB = W + (size_t)(n0 + lr) * K + lsg;

    const v8f zero8 = {0.f, 0.f, 0.f, 0.f, 0.f, 0.f, 0.f, 0.f};
    v8f acc[2][4];
#pragma unroll
    for (int mt = 0; mt < 2; ++mt)
#pragma unroll
        for (int nt = 0; nt < 4; ++nt) acc[mt][nt] = zero8;

    const int NSTAGE = K >> 5;

    // prologue: stage 0
    {
        async_copy_b128(lds_byte_off(&As[0][lr][lsg]),     gA);
        async_copy_b128(lds_byte_off(&As[0][lr][lsg + 8]), gA + 8);
        async_copy_b128(lds_byte_off(&Bs[0][lr][lsg]),     gB);
        async_copy_b128(lds_byte_off(&Bs[0][lr][lsg + 8]), gB + 8);
    }

    for (int s = 0; s < NSTAGE; ++s) {
        wait_async0();        // own stage-s async writes have landed
        __syncthreads();      // everyone's stage-s data visible; stage s-1 reads done

        if (s + 1 < NSTAGE) { // overlap stage s+1 copy with stage-s WMMAs
            const int nb = (s + 1) & 1;
            const int k1 = (s + 1) << 5;
            async_copy_b128(lds_byte_off(&As[nb][lr][lsg]),     gA + k1);
            async_copy_b128(lds_byte_off(&As[nb][lr][lsg + 8]), gA + k1 + 8);
            async_copy_b128(lds_byte_off(&Bs[nb][lr][lsg]),     gB + k1);
            async_copy_b128(lds_byte_off(&Bs[nb][lr][lsg + 8]), gB + k1 + 8);
        }

        const int buf = s & 1;
        FragBF a[2], b[4];
#pragma unroll
        for (int mt = 0; mt < 2; ++mt) {          // A: lane = M row, K chunks 8+8
            int m = wm + mt * 16 + ln, base = lh * 8;
            a[mt].f[0] = *(const f32x4*)&As[buf][m][base];
            a[mt].f[1] = *(const f32x4*)&As[buf][m][base + 16];
        }
#pragma unroll
        for (int nt = 0; nt < 4; ++nt) {          // B: lane = N col, K chunks 8+8
            int n = wn + nt * 16 + ln, kb = lh * 8;
            b[nt].f[0] = *(const f32x4*)&Bs[buf][n][kb];
            b[nt].f[1] = *(const f32x4*)&Bs[buf][n][kb + 16];
        }
#pragma unroll
        for (int mt = 0; mt < 2; ++mt)
#pragma unroll
            for (int nt = 0; nt < 4; ++nt)
                acc[mt][nt] = __builtin_amdgcn_wmma_f32_16x16x32_bf16(
                    false, a[mt].v, false, b[nt].v, (short)0, acc[mt][nt], false, false);
    }

    const float sc = scale_mul * (scale_ptr ? *scale_ptr : 1.0f);
#pragma unroll
    for (int mt = 0; mt < 2; ++mt)
#pragma unroll
        for (int nt = 0; nt < 4; ++nt) {
            int col = n0 + wn + nt * 16 + ln;
            float bv = bias ? bias[col] : 0.0f;
#pragma unroll
            for (int r = 0; r < 8; ++r) {
                int row = m0 + wm + mt * 16 + lh * 8 + r;
                float v = (acc[mt][nt][r] + bv) * sc;
                if (GELU) v = gelu_exact(v);
                if (OUTBF) Cb[(size_t)row * N + col] = (bf16)v;
                else       Cf[(size_t)row * N + col] = v;
            }
        }
}

// ---------------------------------------------------------------- attention
// One wave per (b, h, 16-query tile). Dynamic LDS:
//   float sc[16][SEQ]  (128 KB score/prob strip)  + bf16 stage[2048] (K/V tiles)
__global__ __launch_bounds__(32)
void attn_topk_kernel(const bf16* __restrict__ Q, const bf16* __restrict__ Kb,
                      const bf16* __restrict__ Vb, bf16* __restrict__ CTX) {
    extern __shared__ char smem_raw[];
    float* sc   = (float*)smem_raw;                       // 16 * SEQ floats
    bf16* stage = (bf16*)(smem_raw + 16 * SEQ * sizeof(float));  // 2048 bf16

    const int lane = threadIdx.x;
    const int qt = blockIdx.x, h = blockIdx.y, b = blockIdx.z;
    const int q0 = qt * 16;
    const int ln = lane & 15, lh = lane >> 4;
    const size_t headbase = (size_t)b * SEQ * DMODEL + (size_t)h * DK;

    const v8f zero8 = {0.f, 0.f, 0.f, 0.f, 0.f, 0.f, 0.f, 0.f};

    // ---- Q fragments (dk 0..31, dk 32..63), loaded once
    FragBF aq0, aq1;
    {
        size_t qrow = headbase + (size_t)(q0 + ln) * DMODEL;
        int base = lh * 8;
        aq0.f[0] = *(const f32x4*)&Q[qrow + base];
        aq0.f[1] = *(const f32x4*)&Q[qrow + base + 16];
        aq1.f[0] = *(const f32x4*)&Q[qrow + 32 + base];
        aq1.f[1] = *(const f32x4*)&Q[qrow + 32 + base + 16];
    }

    // ---- scores = Q K^T / sqrt(DK), causal-masked, into LDS strip
    for (int kt = 0; kt <= qt; ++kt) {
        { // stage K tile 16x64 row-major (each lane: half a row, 64B)
            int row = ln, half = lh * 32;
            const f32x4* src = (const f32x4*)&Kb[headbase + (size_t)(kt * 16 + row) * DMODEL + half];
            f32x4* dst = (f32x4*)&stage[row * 64 + half];
            dst[0] = src[0]; dst[1] = src[1]; dst[2] = src[2]; dst[3] = src[3];
        }
        __syncthreads();
        FragBF bk0, bk1;
        {
            int n = ln, kb = lh * 8;
            bk0.f[0] = *(const f32x4*)&stage[n * 64 + kb];
            bk0.f[1] = *(const f32x4*)&stage[n * 64 + kb + 16];
            bk1.f[0] = *(const f32x4*)&stage[n * 64 + 32 + kb];
            bk1.f[1] = *(const f32x4*)&stage[n * 64 + 32 + kb + 16];
        }
        v8f c = zero8;
        c = __builtin_amdgcn_wmma_f32_16x16x32_bf16(false, aq0.v, false, bk0.v, (short)0, c, false, false);
        c = __builtin_amdgcn_wmma_f32_16x16x32_bf16(false, aq1.v, false, bk1.v, (short)0, c, false, false);
#pragma unroll
        for (int r = 0; r < 8; ++r) {
            int row = lh * 8 + r;           // local query row
            int col = kt * 16 + ln;         // key index
            float v = c[r] * 0.125f;        // 1/sqrt(64)
            if (col > q0 + row) v = NEGV;   // causal
            sc[row * SEQ + col] = v;
        }
        __syncthreads();
    }

    const int kcount = (qt + 1) * 16;

    // ---- per-row top-8 threshold + softmax (one lane per query row)
    if (lane < 16) {
        float* srow = sc + lane * SEQ;
        float t[8];
#pragma unroll
        for (int j = 0; j < 8; ++j) t[j] = -3.0e38f;
        for (int i = 0; i < kcount; ++i) {
            float x = srow[i];
            if (x > t[7]) {
                t[7] = x;
#pragma unroll
                for (int j = 7; j > 0; --j)
                    if (t[j] > t[j - 1]) { float tmp = t[j - 1]; t[j - 1] = t[j]; t[j] = tmp; }
            }
        }
        float thr = t[7];
        float mx = -3.0e38f;
        for (int i = 0; i < kcount; ++i) {
            float x = srow[i];
            if (x >= thr && x > mx) mx = x;
        }
        float sum = 0.0f;
        for (int i = 0; i < kcount; ++i) {
            float x = srow[i];
            float e = (x >= thr) ? __expf(x - mx) : 0.0f;
            srow[i] = e;
            sum += e;
        }
        float inv = 1.0f / sum;
        for (int i = 0; i < kcount; ++i) srow[i] *= inv;
        if (kcount & 31)                       // zero-pad to 32-key boundary
            for (int i = kcount; i < kcount + 16; ++i) srow[i] = 0.0f;
    }
    __syncthreads();

    // ---- ctx = P @ V   (K-chunks of 32 keys, N = 64 dk as 4 tiles of 16)
    v8f accn[4];
    accn[0] = zero8; accn[1] = zero8; accn[2] = zero8; accn[3] = zero8;
    const int nch = (kcount + 31) / 32;
    for (int kc = 0; kc < nch; ++kc) {
        { // stage V chunk transposed: vt[dk][key]  (64 x 32)
            int key = lane;
            bf16 tmp[64];
            const f32x4* src = (const f32x4*)&Vb[headbase + (size_t)(kc * 32 + key) * DMODEL];
            f32x4* t4 = (f32x4*)tmp;
#pragma unroll
            for (int j = 0; j < 8; ++j) t4[j] = src[j];
#pragma unroll
            for (int d = 0; d < 64; ++d) stage[d * 32 + key] = tmp[d];
        }
        __syncthreads();

        FragBF ap;   // probabilities -> bf16 A fragment
        {
            const float* pr = sc + ln * SEQ + kc * 32;
            int base = lh * 8;
#pragma unroll
            for (int j = 0; j < 8; ++j) {
                ap.e[j]     = (bf16)pr[base + j];
                ap.e[8 + j] = (bf16)pr[base + 16 + j];
            }
        }
#pragma unroll
        for (int nt = 0; nt < 4; ++nt) {
            FragBF bv;
            int d = nt * 16 + ln, kb = lh * 8;
            bv.f[0] = *(const f32x4*)&stage[d * 32 + kb];
            bv.f[1] = *(const f32x4*)&stage[d * 32 + kb + 16];
            accn[nt] = __builtin_amdgcn_wmma_f32_16x16x32_bf16(
                false, ap.v, false, bv.v, (short)0, accn[nt], false, false);
        }
        __syncthreads();
    }

    // ---- store ctx (bf16, [B,S,D] with head offset)
#pragma unroll
    for (int nt = 0; nt < 4; ++nt)
#pragma unroll
        for (int r = 0; r < 8; ++r) {
            int row = q0 + lh * 8 + r;
            int d = nt * 16 + ln;
            CTX[(size_t)b * SEQ * DMODEL + (size_t)row * DMODEL + h * DK + d] = (bf16)accn[nt][r];
        }
}

// ---------------------------------------------------------------- launch
extern "C" void kernel_launch(void* const* d_in, const int* in_sizes, int n_in,
                              void* d_out, int out_size, void* d_ws, size_t ws_size,
                              hipStream_t stream) {
    const float* x  = (const float*)d_in[0];
    const float* Wq = (const float*)d_in[1];
    const float* bq = (const float*)d_in[2];
    const float* Wk = (const float*)d_in[3];
    const float* bk = (const float*)d_in[4];
    const float* Wv = (const float*)d_in[5];
    const float* bv = (const float*)d_in[6];
    const float* Wo = (const float*)d_in[7];
    const float* bo = (const float*)d_in[8];
    const float* g1 = (const float*)d_in[9];
    const float* W1 = (const float*)d_in[10];
    const float* b1 = (const float*)d_in[11];
    const float* W2 = (const float*)d_in[12];
    const float* b2 = (const float*)d_in[13];
    const float* g2 = (const float*)d_in[14];
    float* out = (float*)d_out;

    // workspace layout (bf16 buffers, with reuse)
    char* ws = (char*)d_ws;
    const size_t SZ_W   = (size_t)DFF * DMODEL * sizeof(bf16);      // 8 MB (max weight)
    const size_t SZ_ACT = (size_t)MROWS * DMODEL * sizeof(bf16);    // 8 MB
    bf16* wbuf = (bf16*)(ws);                                       // current weight (bf16)
    bf16* xb   = (bf16*)(ws + SZ_W);                                // x bf16 / later ctx
    bf16* qb   = (bf16*)(ws + SZ_W + SZ_ACT);                       // q / later h
    bf16* kbuf = (bf16*)(ws + SZ_W + 2 * SZ_ACT);
    bf16* vbuf = (bf16*)(ws + SZ_W + 3 * SZ_ACT);
    bf16* gb   = (bf16*)(ws + SZ_W + 4 * SZ_ACT);                   // gelu out (32 MB)
    bf16* ctxb = xb;
    bf16* hb   = qb;

    const long nAct = (long)MROWS * DMODEL;       // 4M
    const long nWsq = (long)DMODEL * DMODEL;      // 1M
    const long nWff = (long)DFF * DMODEL;         // 4M
    dim3 cvtGrid(2048), cvtBlk(256);

    dim3 gB(256);
    dim3 gridD(DMODEL / 128, MROWS / 128);        // N=1024 GEMMs: (8, 32)
    dim3 gridF(DFF / 128, MROWS / 128);           // N=4096 GEMM:  (32, 32)

    // x -> bf16
    cvt_f32_bf16<<<cvtGrid, cvtBlk, 0, stream>>>(x, xb, nAct);

    // Q = x Wq^T + bq
    cvt_f32_bf16<<<cvtGrid, cvtBlk, 0, stream>>>(Wq, wbuf, nWsq);
    gemm_bf16_nt<false, true><<<gridD, gB, 0, stream>>>(xb, wbuf, bq, nullptr, 1.0f,
                                                        nullptr, qb, MROWS, DMODEL, DMODEL);
    // K
    cvt_f32_bf16<<<cvtGrid, cvtBlk, 0, stream>>>(Wk, wbuf, nWsq);
    gemm_bf16_nt<false, true><<<gridD, gB, 0, stream>>>(xb, wbuf, bk, nullptr, 1.0f,
                                                        nullptr, kbuf, MROWS, DMODEL, DMODEL);
    // V
    cvt_f32_bf16<<<cvtGrid, cvtBlk, 0, stream>>>(Wv, wbuf, nWsq);
    gemm_bf16_nt<false, true><<<gridD, gB, 0, stream>>>(xb, wbuf, bv, nullptr, 1.0f,
                                                        nullptr, vbuf, MROWS, DMODEL, DMODEL);

    // attention (top-8 sparsified causal softmax) -> ctx (overwrites xb)
    dim3 aGrid(SEQ / 16, NHEAD, BATCH), aBlk(32);
    size_t attnLds = 16 * SEQ * sizeof(float) + 2048 * sizeof(bf16);  // 135168 B
    attn_topk_kernel<<<aGrid, aBlk, attnLds, stream>>>(qb, kbuf, vbuf, ctxb);

    // h = 2*g1*(ctx Wo^T + bo)   (overwrites qb)
    cvt_f32_bf16<<<cvtGrid, cvtBlk, 0, stream>>>(Wo, wbuf, nWsq);
    gemm_bf16_nt<false, true><<<gridD, gB, 0, stream>>>(ctxb, wbuf, bo, g1, 2.0f,
                                                        nullptr, hb, MROWS, DMODEL, DMODEL);
    // g = gelu(h W1^T + b1)
    cvt_f32_bf16<<<cvtGrid, cvtBlk, 0, stream>>>(W1, wbuf, nWff);
    gemm_bf16_nt<true, true><<<gridF, gB, 0, stream>>>(hb, wbuf, b1, nullptr, 1.0f,
                                                       nullptr, gb, MROWS, DFF, DMODEL);
    // y = 2*g2*(g W2^T + b2)  -> f32 output
    cvt_f32_bf16<<<cvtGrid, cvtBlk, 0, stream>>>(W2, wbuf, nWff);
    gemm_bf16_nt<false, false><<<gridD, gB, 0, stream>>>(gb, wbuf, b2, g2, 2.0f,
                                                         out, nullptr, MROWS, DMODEL, DFF);
}
